// _ReformerCore_18966575579404
// MI455X (gfx1250) — matrix operand order repaired and checked by reading, not compiled, not run on hardware
//
#include <hip/hip_runtime.h>
#include <hip/hip_bf16.h>
#include <math.h>

// ---------------- model constants ----------------
#define BB 8
#define LL 2048
#define DD 512
#define HH 8
#define DHH 64
#define BHH 64        // BB*HH
#define NHR 4         // hash rounds
#define NBK 512       // buckets per round
#define DFF_ 2048
#define MM 16384      // BB*LL
#define ENC_ 21
#define COUT_ 21

typedef _Float16 half_t;
typedef half_t v16h __attribute__((ext_vector_type(16)));
typedef half_t h8   __attribute__((ext_vector_type(8)));
typedef half_t h4   __attribute__((ext_vector_type(4)));
typedef float  v8f  __attribute__((ext_vector_type(8)));

// ---------------- helpers ----------------
__device__ inline float wred(float x) {
#pragma unroll
    for (int off = 16; off > 0; off >>= 1) x += __shfl_xor(x, off, 32);
    return x;
}

__device__ inline float gelu_exact(float x) {
    return 0.5f * x * (1.f + erff(x * 0.70710678118654752f));
}

// ---------------- embedding ----------------
__global__ void embed_kernel(const float* __restrict__ x_enc, const float* __restrict__ x_mark,
                             const float* __restrict__ conv_w, const float* __restrict__ temp_w,
                             float* __restrict__ X) {
    int idx = blockIdx.x * 256 + threadIdx.x;           // over 8*2048*512
    int d = idx & 511;
    int n = (idx >> 9) & 2047;
    int b = idx >> 20;
    float acc = 0.f;
#pragma unroll
    for (int k = 0; k < 3; ++k) {
        int lm = (n + k + LL - 1) & (LL - 1);
        const float* xe = x_enc + ((size_t)b * LL + lm) * ENC_;
        const float* w  = conv_w + (size_t)d * (ENC_ * 3) + k;
        for (int c = 0; c < ENC_; ++c) acc += xe[c] * w[c * 3];
    }
    int i2 = (d >> 1) * 2;
    float div = __expf(-(float)i2 * (9.210340371976184f / (float)DD)); // ln(10000)
    float pe = (d & 1) ? __cosf((float)n * div) : __sinf((float)n * div);
    const float* xm = x_mark + ((size_t)b * LL + n) * 4;
    float tmp = 0.f;
#pragma unroll
    for (int m = 0; m < 4; ++m) tmp += xm[m] * temp_w[m * DD + d];
    X[idx] = acc + pe + tmp;
}

// ---------------- WMMA GEMM ----------------
// C[M,N] = epilogue(A[M,K] @ B + bias).  TRANSB==0: B is [K,N].  TRANSB==1: B is [N,K].
// HASGELU: apply exact GELU in the epilogue (fused activation, saves a 268MB pass).
// Block tile 128x128, K-step 32, double-buffered LDS, 8 wave32s (4x2), wave: 32x64 (2x4 WMMA).
template <int TRANSB, int HASBIAS, int HASGELU>
__global__ __launch_bounds__(256) void gemm_wmma(const float* __restrict__ A,
                                                 const float* __restrict__ Bm,
                                                 const float* __restrict__ bias,
                                                 float* __restrict__ C,
                                                 int Ndim, int Kdim) {
    __shared__ half_t As[2][128][40];
    __shared__ half_t Bs[2][128][40];
    int tid  = threadIdx.x;
    int lane = tid & 31;
    int wave = tid >> 5;
    int wm = wave >> 1, wn = wave & 1;
    int row0 = blockIdx.y * 128, col0 = blockIdx.x * 128;
    int lr = lane & 15, lh = lane >> 4;

    v8f acc[2][4];
#pragma unroll
    for (int fm = 0; fm < 2; ++fm)
#pragma unroll
        for (int fn = 0; fn < 4; ++fn) acc[fm][fn] = (v8f){};
    int nk = Kdim >> 5;

    auto stage = [&](int kb, int buf) {
        int k0 = kb << 5;
        float4 av[4], bv[4];
#pragma unroll
        for (int t = 0; t < 4; ++t) {
            int idx = t * 256 + tid;
            av[t] = *(const float4*)(A + (size_t)(row0 + (idx >> 3)) * Kdim + k0 + ((idx & 7) << 2));
        }
        if (TRANSB) {
#pragma unroll
            for (int t = 0; t < 4; ++t) {
                int idx = t * 256 + tid;
                bv[t] = *(const float4*)(Bm + (size_t)(col0 + (idx >> 3)) * Kdim + k0 + ((idx & 7) << 2));
            }
        } else {
#pragma unroll
            for (int t = 0; t < 4; ++t) {
                int idx = t * 256 + tid;
                bv[t] = *(const float4*)(Bm + (size_t)(k0 + (idx >> 5)) * Ndim + col0 + ((idx & 31) << 2));
            }
        }
#pragma unroll
        for (int t = 0; t < 4; ++t) {
            int idx = t * 256 + tid;
            h4 hv; hv[0] = (half_t)av[t].x; hv[1] = (half_t)av[t].y;
            hv[2] = (half_t)av[t].z; hv[3] = (half_t)av[t].w;
            *(h4*)&As[buf][idx >> 3][(idx & 7) << 2] = hv;
        }
        if (TRANSB) {
#pragma unroll
            for (int t = 0; t < 4; ++t) {
                int idx = t * 256 + tid;
                h4 hv; hv[0] = (half_t)bv[t].x; hv[1] = (half_t)bv[t].y;
                hv[2] = (half_t)bv[t].z; hv[3] = (half_t)bv[t].w;
                *(h4*)&Bs[buf][idx >> 3][(idx & 7) << 2] = hv;
            }
        } else {
#pragma unroll
            for (int t = 0; t < 4; ++t) {
                int idx = t * 256 + tid;
                int i = idx >> 5, j4 = (idx & 31) << 2;
                Bs[buf][j4 + 0][i] = (half_t)bv[t].x;
                Bs[buf][j4 + 1][i] = (half_t)bv[t].y;
                Bs[buf][j4 + 2][i] = (half_t)bv[t].z;
                Bs[buf][j4 + 3][i] = (half_t)bv[t].w;
            }
        }
    };

    stage(0, 0);
    __syncthreads();

    for (int kb = 0; kb < nk; ++kb) {
        int cur = kb & 1;
        if (kb + 1 < nk) stage(kb + 1, cur ^ 1);           // overlap staging with WMMA
        if (kb + 2 < nk) {                                  // global_prefetch for tile kb+2
            __builtin_prefetch(A + (size_t)(row0 + (tid >> 1)) * Kdim + ((kb + 2) << 5), 0, 1);
        }
        v16h a[2], bf[4];
#pragma unroll
        for (int fm = 0; fm < 2; ++fm) {
            int r = wm * 32 + fm * 16 + lr;
            a[fm] = __builtin_shufflevector(*(const h8*)&As[cur][r][lh * 8],
                                            *(const h8*)&As[cur][r][16 + lh * 8],
                                            0,1,2,3,4,5,6,7,8,9,10,11,12,13,14,15);
        }
#pragma unroll
        for (int fn = 0; fn < 4; ++fn) {
            int cb = wn * 64 + fn * 16 + lr;
            bf[fn] = __builtin_shufflevector(*(const h8*)&Bs[cur][cb][lh * 8],
                                             *(const h8*)&Bs[cur][cb][16 + lh * 8],
                                             0,1,2,3,4,5,6,7,8,9,10,11,12,13,14,15);
        }
#pragma unroll
        for (int fm = 0; fm < 2; ++fm)
#pragma unroll
            for (int fn = 0; fn < 4; ++fn)
                acc[fm][fn] = __builtin_amdgcn_wmma_f32_16x16x32_f16(
                    false, a[fm], false, bf[fn], (short)0, acc[fm][fn], false, false);
        __syncthreads();
    }

    int gr = row0 + wm * 32 + lh * 8;
#pragma unroll
    for (int fn = 0; fn < 4; ++fn) {
        int gc = col0 + wn * 64 + fn * 16 + lr;
        float bb = HASBIAS ? bias[gc] : 0.f;
        float* cp = C + (size_t)gr * Ndim + gc;
#pragma unroll
        for (int r = 0; r < 8; ++r) {
            float v0 = acc[0][fn][r] + bb;
            float v1 = acc[1][fn][r] + bb;
            if (HASGELU) { v0 = gelu_exact(v0); v1 = gelu_exact(v1); }
            cp[(size_t)r * Ndim]        = v0;
            cp[(size_t)(r + 16) * Ndim] = v1;
        }
    }
}

// ---------------- LSH bucket assignment via WMMA ----------------
// rotated = qk[131072,64] @ rot[64,1024]; block = 64 rows x 256 cols (one full hash group)
// so argmax over [rot,-rot] (512 candidates) finishes inside the block.
__global__ __launch_bounds__(256) void lsh_buckets_wmma(const float* __restrict__ QK,
                                                        const float* __restrict__ rot,
                                                        int* __restrict__ buck) {
    __shared__ half_t As[64][72];     // pitch 72 halves = 144B (16B multiple)
    __shared__ half_t Bsh[256][72];
    __shared__ float redv[4][64];
    __shared__ int   redi[4][64];
    int tid = threadIdx.x, lane = tid & 31, wave = tid >> 5;
    int wm = wave >> 2, wn = wave & 3;
    int lr = lane & 15, lh = lane >> 4;
    int hg = blockIdx.x;                       // hash round 0..3
    int rr0 = blockIdx.y * 64;                 // 2048 row-blocks over BH*L rows
    int bh = rr0 >> 11, n0 = rr0 & 2047;
    int b = bh >> 3, hd = bh & 7;

    // stage A: 64 rows x 64 k of qk (f32 -> f16)
    const float* Abase = QK + ((size_t)(b * LL + n0) * DD + hd * DHH);
#pragma unroll
    for (int t = 0; t < 4; ++t) {
        int idx = t * 256 + tid;               // 1024 float4
        int i = idx >> 4, j4 = (idx & 15) << 2;
        float4 v = *(const float4*)(Abase + (size_t)i * DD + j4);
        h4 hv; hv[0] = (half_t)v.x; hv[1] = (half_t)v.y; hv[2] = (half_t)v.z; hv[3] = (half_t)v.w;
        *(h4*)&As[i][j4] = hv;
    }
    // stage B: rot[d][hg*256 + c] -> Bsh[c][d] (N-major)
    const float* Rbase = rot + hg * 256;
#pragma unroll
    for (int t = 0; t < 16; ++t) {
        int idx = t * 256 + tid;               // 4096 float4
        int d = idx >> 6, c4 = (idx & 63) << 2;
        float4 v = *(const float4*)(Rbase + (size_t)d * 1024 + c4);
        Bsh[c4 + 0][d] = (half_t)v.x;
        Bsh[c4 + 1][d] = (half_t)v.y;
        Bsh[c4 + 2][d] = (half_t)v.z;
        Bsh[c4 + 3][d] = (half_t)v.w;
    }
    __syncthreads();

    v8f acc[2][4];
#pragma unroll
    for (int fm = 0; fm < 2; ++fm)
#pragma unroll
        for (int fn = 0; fn < 4; ++fn) acc[fm][fn] = (v8f){};

#pragma unroll
    for (int ks = 0; ks < 2; ++ks) {
        int kk = ks * 32;
        v16h a[2], bf[4];
#pragma unroll
        for (int fm = 0; fm < 2; ++fm) {
            int r = wm * 32 + fm * 16 + lr;
            a[fm] = __builtin_shufflevector(*(const h8*)&As[r][kk + lh * 8],
                                            *(const h8*)&As[r][kk + 16 + lh * 8],
                                            0,1,2,3,4,5,6,7,8,9,10,11,12,13,14,15);
        }
#pragma unroll
        for (int fn = 0; fn < 4; ++fn) {
            int cb = wn * 64 + fn * 16 + lr;
            bf[fn] = __builtin_shufflevector(*(const h8*)&Bsh[cb][kk + lh * 8],
                                             *(const h8*)&Bsh[cb][kk + 16 + lh * 8],
                                             0,1,2,3,4,5,6,7,8,9,10,11,12,13,14,15);
        }
#pragma unroll
        for (int fm = 0; fm < 2; ++fm)
#pragma unroll
            for (int fn = 0; fn < 4; ++fn)
                acc[fm][fn] = __builtin_amdgcn_wmma_f32_16x16x32_f16(
                    false, a[fm], false, bf[fn], (short)0, acc[fm][fn], false, false);
    }

    // argmax epilogue: lane-local over 4 col frags (+/-), then 16-lane half-wave reduce
#pragma unroll
    for (int fm = 0; fm < 2; ++fm) {
#pragma unroll
        for (int r = 0; r < 8; ++r) {
            float bvv = -3.4e38f; int bii = 0;
#pragma unroll
            for (int fn = 0; fn < 4; ++fn) {
                float v = acc[fm][fn][r];
                int c = wn * 64 + fn * 16 + lr;
                if (v > bvv)  { bvv = v;  bii = c; }
                if (-v > bvv) { bvv = -v; bii = c + 256; }
            }
#pragma unroll
            for (int m = 1; m < 16; m <<= 1) {
                float ov = __shfl_xor(bvv, m, 32);
                int   oi = __shfl_xor(bii, m, 32);
                if (ov > bvv || (ov == bvv && oi < bii)) { bvv = ov; bii = oi; }
            }
            if (lr == 0) {
                int row = wm * 32 + fm * 16 + lh * 8 + r;
                redv[wn][row] = bvv;
                redi[wn][row] = bii;
            }
        }
    }
    __syncthreads();
    if (tid < 64) {
        float bvv = -3.4e38f; int bii = 0;
#pragma unroll
        for (int wni = 0; wni < 4; ++wni) {
            float v = redv[wni][tid]; int i = redi[wni][tid];
            if (v > bvv || (v == bvv && i < bii)) { bvv = v; bii = i; }
        }
        buck[(size_t)bh * (NHR * LL) + hg * LL + (n0 + tid)] = bii + hg * NBK;
    }
}

// ---------------- stable counting sort per (b,head) ----------------
__global__ __launch_bounds__(256) void lsh_sort(const int* __restrict__ buck,
                                                int* __restrict__ sorted) {
    __shared__ int hist[2048];
    __shared__ int part[256];
    int tid = threadIdx.x, bh = blockIdx.x;
    for (int t = tid; t < 2048; t += 256) hist[t] = 0;
    __syncthreads();
    for (int t = tid; t < 8192; t += 256)
        atomicAdd(&hist[buck[(size_t)bh * 8192 + t]], 1);
    __syncthreads();
    int base = tid * 8, local = 0;
#pragma unroll
    for (int i = 0; i < 8; ++i) local += hist[base + i];
    part[tid] = local;
    __syncthreads();
    for (int st = 1; st < 256; st <<= 1) {
        int v = (tid >= st) ? part[tid - st] : 0;
        __syncthreads();
        part[tid] += v;
        __syncthreads();
    }
    int run = part[tid] - local;
#pragma unroll
    for (int i = 0; i < 8; ++i) { int c = hist[base + i]; hist[base + i] = run; run += c; }
    __syncthreads();
    if (tid < 4) {                            // bins disjoint across hash rounds -> race-free
        int h = tid;
        for (int n = 0; n < 2048; ++n) {
            int bk = buck[(size_t)bh * 8192 + h * 2048 + n];
            int p = hist[bk]++;
            sorted[(size_t)bh * 8192 + p] = h * 2048 + n;
        }
    }
}

// ---------------- chunked LSH attention (one wave per (bh, chunk)) ----------------
__global__ __launch_bounds__(256) void lsh_attn(const float* __restrict__ QK,
                                                const float* __restrict__ V,
                                                const int* __restrict__ sorted,
                                                float* __restrict__ oh,
                                                float* __restrict__ logits) {
    int tid = threadIdx.x, lane = tid & 31;
    int w = blockIdx.x * 8 + (tid >> 5);      // 131072 waves = BH * 2048 chunks
    int bh = w >> 11, c = w & 2047;
    int b = bh >> 3, hd = bh & 7;
    const int* sp = sorted + (size_t)bh * 8192;

    int tq[4], nq[4];
    float2 qv[4];
#pragma unroll
    for (int i = 0; i < 4; ++i) {
        tq[i] = sp[c * 4 + i];
        nq[i] = tq[i] & 2047;
        const float2* qp = (const float2*)(QK + ((size_t)(b * LL + nq[i]) * DD + hd * DHH));
        qv[i] = qp[lane];
    }
    int cm1 = (c + 2047) & 2047;              // look one chunk back (roll)
    int nk[8];
    float dots[4][8];
#pragma unroll
    for (int j = 0; j < 8; ++j) {
        int s = (j < 4) ? (c * 4 + j) : (cm1 * 4 + (j - 4));
        int tk = sp[s];
        nk[j] = tk & 2047;
        const float2* kp = (const float2*)(QK + ((size_t)(b * LL + nk[j]) * DD + hd * DHH));
        float2 kv = kp[lane];
        float nrm = sqrtf(wred(kv.x * kv.x + kv.y * kv.y));
        float inv = 0.125f / fmaxf(nrm, 1e-12f);   // DH^-0.5 / ||k||
#pragma unroll
        for (int i = 0; i < 4; ++i) {
            float dd = wred(qv[i].x * kv.x + qv[i].y * kv.y) * inv;
            dots[i][j] = (nq[i] == nk[j]) ? -5e4f : dd;   // mask self
        }
    }
    float out[4][2] = {};
    float lse[4], pj[4][8];
#pragma unroll
    for (int i = 0; i < 4; ++i) {
        float m = dots[i][0];
#pragma unroll
        for (int j = 1; j < 8; ++j) m = fmaxf(m, dots[i][j]);
        float s = 0.f;
#pragma unroll
        for (int j = 0; j < 8; ++j) { pj[i][j] = __expf(dots[i][j] - m); s += pj[i][j]; }
        float inv = 1.f / s;
#pragma unroll
        for (int j = 0; j < 8; ++j) pj[i][j] *= inv;
        lse[i] = m + __logf(s);
    }
#pragma unroll
    for (int j = 0; j < 8; ++j) {
        const float2* vp = (const float2*)(V + ((size_t)(b * LL + nk[j]) * DD + hd * DHH));
        float2 vv = vp[lane];
#pragma unroll
        for (int i = 0; i < 4; ++i) { out[i][0] += pj[i][j] * vv.x; out[i][1] += pj[i][j] * vv.y; }
    }
#pragma unroll
    for (int i = 0; i < 4; ++i) {             // scatter through permutation: no undo pass needed
        int h = tq[i] >> 11;
        size_t off = (((size_t)bh * NHR + h) * LL + nq[i]) * DHH;
        float2* op = (float2*)(oh + off);
        float2 o2; o2.x = out[i][0]; o2.y = out[i][1];
        op[lane] = o2;
        if (lane == 0) logits[((size_t)bh * NHR + h) * LL + nq[i]] = lse[i];
    }
}

// ---------------- combine hash rounds (softmax over lse) -> [B,L,D] ----------------
__global__ void lsh_combine(const float* __restrict__ oh, const float* __restrict__ logits,
                            float* __restrict__ oc) {
    int idx = blockIdx.x * 256 + threadIdx.x; // over 8*2048*512
    int d = idx & 511;
    int n = (idx >> 9) & 2047;
    int b = idx >> 20;
    int hd = d >> 6, dh = d & 63;
    int bh = b * HH + hd;
    float l[4], m = -3.4e38f;
#pragma unroll
    for (int h = 0; h < 4; ++h) {
        l[h] = logits[((size_t)bh * NHR + h) * LL + n];
        m = fmaxf(m, l[h]);
    }
    float s = 0.f;
#pragma unroll
    for (int h = 0; h < 4; ++h) { l[h] = __expf(l[h] - m); s += l[h]; }
    float inv = 1.f / s, acc = 0.f;
#pragma unroll
    for (int h = 0; h < 4; ++h)
        acc += oh[(((size_t)bh * NHR + h) * LL + n) * DHH + dh] * (l[h] * inv);
    oc[idx] = acc;
}

// ---------------- residual + layernorm (row = one block) ----------------
__global__ __launch_bounds__(256) void ln_kernel(const float* __restrict__ inp,
                                                 const float* __restrict__ resid,
                                                 const float* __restrict__ g,
                                                 const float* __restrict__ bta,
                                                 float* __restrict__ out) {
    int row = blockIdx.x, t = threadIdx.x;
    const float* ip = inp + (size_t)row * DD;
    float v0 = ip[t]       + (resid ? resid[(size_t)row * DD + t]       : 0.f);
    float v1 = ip[t + 256] + (resid ? resid[(size_t)row * DD + t + 256] : 0.f);
    __shared__ float s1[256], s2[256];
    s1[t] = v0 + v1;
    s2[t] = v0 * v0 + v1 * v1;
    __syncthreads();
    for (int st = 128; st > 0; st >>= 1) {
        if (t < st) { s1[t] += s1[t + st]; s2[t] += s2[t + st]; }
        __syncthreads();
    }
    float mean = s1[0] * (1.f / DD);
    float var  = s2[0] * (1.f / DD) - mean * mean;
    float rstd = rsqrtf(var + 1e-5f);
    out[(size_t)row * DD + t]       = (v0 - mean) * rstd * g[t] + bta[t];
    out[(size_t)row * DD + t + 256] = (v1 - mean) * rstd * g[t + 256] + bta[t + 256];
}

// ---------------- final projection D -> 21 ----------------
__global__ void proj_kernel(const float* __restrict__ Xn, const float* __restrict__ W,
                            const float* __restrict__ bias, float* __restrict__ out) {
    int idx = blockIdx.x * 256 + threadIdx.x;
    if (idx >= MM * COUT_) return;
    int row = idx / COUT_, c = idx % COUT_;
    const float* xp = Xn + (size_t)row * DD;
    float acc = bias[c];
    for (int d = 0; d < DD; ++d) acc += xp[d] * W[d * COUT_ + c];
    out[idx] = acc;
}

// ---------------- launch ----------------
extern "C" void kernel_launch(void* const* d_in, const int* in_sizes, int n_in,
                              void* d_out, int out_size, void* d_ws, size_t ws_size,
                              hipStream_t stream) {
    (void)in_sizes; (void)n_in; (void)out_size; (void)ws_size;
    const float* x_enc  = (const float*)d_in[0];
    const float* x_mark = (const float*)d_in[1];
    const float* conv_w = (const float*)d_in[2];
    const float* temp_w = (const float*)d_in[3];
    const float* wqk    = (const float*)d_in[4];
    const float* wv     = (const float*)d_in[5];
    const float* wout   = (const float*)d_in[6];
    const float* bout   = (const float*)d_in[7];
    const float* ln1_g  = (const float*)d_in[8];
    const float* ln1_b  = (const float*)d_in[9];
    const float* ln2_g  = (const float*)d_in[10];
    const float* ln2_b  = (const float*)d_in[11];
    const float* ff1_w  = (const float*)d_in[12];
    const float* ff1_b  = (const float*)d_in[13];
    const float* ff2_w  = (const float*)d_in[14];
    const float* ff2_b  = (const float*)d_in[15];
    const float* norm_g = (const float*)d_in[16];
    const float* norm_b = (const float*)d_in[17];
    const float* proj_w = (const float*)d_in[18];
    const float* proj_b = (const float*)d_in[19];
    const float* rot    = (const float*)d_in[20];

    const size_t MD = (size_t)MM * DD;
    float* X     = (float*)d_ws;              // activations [B,L,D]
    float* QK    = X + MD;                    // qk / o_comb / final LN out
    float* Vb    = QK + MD;                   // v
    float* ATT   = Vb + MD;                   // attn_out / ffn_out
    float* BIG   = ATT + MD;                  // ff1 out [M,DFF] == oh [BH,NHR,L,DH] (disjoint lifetimes)
    int*   BUCK  = (int*)(BIG + (size_t)MM * DFF_);
    int*   SORT  = BUCK + (size_t)BHH * NHR * LL;
    float* LOGIT = (float*)(SORT + (size_t)BHH * NHR * LL);

    embed_kernel<<<32768, 256, 0, stream>>>(x_enc, x_mark, conv_w, temp_w, X);

    for (int l = 0; l < 2; ++l) {
        gemm_wmma<0, 0, 0><<<dim3(4, 128), 256, 0, stream>>>(X, wqk + (size_t)l * DD * DD, nullptr, QK, DD, DD);
        gemm_wmma<0, 0, 0><<<dim3(4, 128), 256, 0, stream>>>(X, wv  + (size_t)l * DD * DD, nullptr, Vb, DD, DD);
        lsh_buckets_wmma<<<dim3(4, 2048), 256, 0, stream>>>(QK, rot + (size_t)l * DHH * NHR * (NBK / 2), BUCK);
        lsh_sort<<<64, 256, 0, stream>>>(BUCK, SORT);
        lsh_attn<<<16384, 256, 0, stream>>>(QK, Vb, SORT, BIG, LOGIT);
        lsh_combine<<<32768, 256, 0, stream>>>(BIG, LOGIT, QK);   // QK now holds combined o [B,L,D]
        gemm_wmma<0, 1, 0><<<dim3(4, 128), 256, 0, stream>>>(QK, wout + (size_t)l * DD * DD, bout + (size_t)l * DD, ATT, DD, DD);
        ln_kernel<<<16384, 256, 0, stream>>>(ATT, X, ln1_g + l * DD, ln1_b + l * DD, X);
        gemm_wmma<1, 1, 1><<<dim3(16, 128), 256, 0, stream>>>(X, ff1_w + (size_t)l * DFF_ * DD, ff1_b + (size_t)l * DFF_, BIG, DFF_, DD);   // fused GELU
        gemm_wmma<1, 1, 0><<<dim3(4, 128), 256, 0, stream>>>(BIG, ff2_w + (size_t)l * DD * DFF_, ff2_b + (size_t)l * DD, ATT, DD, DFF_);
        ln_kernel<<<16384, 256, 0, stream>>>(ATT, X, ln2_g + l * DD, ln2_b + l * DD, X);
    }
    ln_kernel<<<16384, 256, 0, stream>>>(X, nullptr, norm_g, norm_b, QK);
    proj_kernel<<<1344, 256, 0, stream>>>(QK, proj_w, proj_b, (float*)d_out);
}
